// FlaxQwen3MoeSparseMoeBlock_85933705659129
// MI455X (gfx1250) — matrix operand order repaired and checked by reading, not compiled, not run on hardware
//
#include <hip/hip_runtime.h>
#include <hip/hip_bf16.h>

typedef __attribute__((ext_vector_type(16))) _Float16 v16h;
typedef __attribute__((ext_vector_type(8)))  _Float16 v8h;
typedef __attribute__((ext_vector_type(8)))  float    v8f;

#define T_TOK 4096
#define DDIM  1024
#define FDIM  1024
#define NEXP  8
#define MTILE 32
#define CAP   4096   // per-expert token capacity (worst case: all tokens)

// pack two f32 -> packed f16 dword with one v_cvt_pk_rtz_f16_f32
__device__ __forceinline__ unsigned int pk16(float a, float b) {
    auto h = __builtin_amdgcn_cvt_pkrtz(a, b);   // __fp16 ext_vector(2)
    return __builtin_bit_cast(unsigned int, h);
}

// ---------------------------------------------------------------------------
// Router: logits, softmax, top-2, renorm; build per-expert token lists.
// One wave (32 lanes) per token.
// ---------------------------------------------------------------------------
__global__ void router_kernel(const float* __restrict__ X,
                              const float* __restrict__ gateW,
                              float* __restrict__ logitsOut,
                              int*   __restrict__ cursor,
                              int*   __restrict__ tokenList,
                              float* __restrict__ wList) {
    const int lane = threadIdx.x & 31;
    const int wave = threadIdx.x >> 5;
    const int t    = blockIdx.x * 8 + wave;

    float acc[NEXP];
#pragma unroll
    for (int e = 0; e < NEXP; ++e) acc[e] = 0.f;

    const float* xrow = X + (size_t)t * DDIM;
#pragma unroll 4
    for (int it = 0; it < DDIM / 32; ++it) {
        const int d = it * 32 + lane;
        const float x  = xrow[d];
        const float4 g0 = *(const float4*)(gateW + d * NEXP);
        const float4 g1 = *(const float4*)(gateW + d * NEXP + 4);
        acc[0] += x * g0.x; acc[1] += x * g0.y; acc[2] += x * g0.z; acc[3] += x * g0.w;
        acc[4] += x * g1.x; acc[5] += x * g1.y; acc[6] += x * g1.z; acc[7] += x * g1.w;
    }
#pragma unroll
    for (int off = 16; off > 0; off >>= 1) {
#pragma unroll
        for (int e = 0; e < NEXP; ++e) acc[e] += __shfl_xor(acc[e], off, 32);
    }

    if (lane == 0) {
        float m = acc[0];
#pragma unroll
        for (int e = 1; e < NEXP; ++e) m = fmaxf(m, acc[e]);
        float p[NEXP]; float s = 0.f;
#pragma unroll
        for (int e = 0; e < NEXP; ++e) { p[e] = __expf(acc[e] - m); s += p[e]; }
        const float inv = 1.f / s;
#pragma unroll
        for (int e = 0; e < NEXP; ++e) {
            p[e] *= inv;
            logitsOut[(size_t)t * NEXP + e] = acc[e];   // raw pre-softmax logits
        }
        int i1 = 0;
#pragma unroll
        for (int e = 1; e < NEXP; ++e) if (p[e] > p[i1]) i1 = e;
        int i2 = -1;
#pragma unroll
        for (int e = 0; e < NEXP; ++e) {
            if (e == i1) continue;
            if (i2 < 0 || p[e] > p[i2]) i2 = e;
        }
        const float denom = 1.f / (p[i1] + p[i2]);
        const float w1 = p[i1] * denom;
        const float w2 = p[i2] * denom;
        int s1 = atomicAdd(&cursor[i1], 1);
        tokenList[i1 * CAP + s1] = t;  wList[i1 * CAP + s1] = w1;
        int s2 = atomicAdd(&cursor[i2], 1);
        tokenList[i2 * CAP + s2] = t;  wList[i2 * CAP + s2] = w2;
    }
}

// ---------------------------------------------------------------------------
// WMMA fragment loaders (layouts per CDNA5 ISA 7.12.2, wave32).
// A (16x32 f16): lane<16 -> row=lane, K = {k0..k0+7, k0+16..k0+23}
//                lane>=16 -> row=lane-16, K = {k0+8..k0+15, k0+24..k0+31}
// B (32x16 f16): lane<16 -> col=lane, K = {k0..k0+7, k0+16..k0+23}
//                lane>=16 -> col=lane-16, K = {k0+8..k0+15, k0+24..k0+31}
// ---------------------------------------------------------------------------
__device__ __forceinline__ v16h load_a(const _Float16* __restrict__ base,
                                       int rowBase, int k0, int lane) {
    const int r  = rowBase + (lane & 15);
    const int kh = (lane >> 4) << 3;            // 0 or 8
    const _Float16* p = base + r * 1024 + k0 + kh;
    const v8h lo = *(const v8h*)p;
    const v8h hi = *(const v8h*)(p + 16);
    v16h a;
#pragma unroll
    for (int i = 0; i < 8; ++i) { a[i] = lo[i]; a[8 + i] = hi[i]; }
    return a;
}

__device__ __forceinline__ v16h load_b(const _Float16* __restrict__ Bsm,  // [128][32]
                                       int colBase, int lane) {
    const int c  = colBase + (lane & 15);
    const int kh = (lane >> 4) << 3;            // 0 or 8
    const v8h lo = *(const v8h*)(Bsm + c * 32 + kh);
    const v8h hi = *(const v8h*)(Bsm + c * 32 + kh + 16);
    v16h b;
#pragma unroll
    for (int i = 0; i < 8; ++i) { b[i] = lo[i]; b[8 + i] = hi[i]; }
    return b;
}

// Stage a [k0..k0+32) x [ncB..ncB+128) fp32 weight tile, transposed, as f16
// into dst viewed as [128 cols][16 k-pairs] dwords. 2K x 4F units -> packed
// cvt + ds_store_b32 (no scalar b16 scatter).
__device__ __forceinline__ void stage_w(const float* __restrict__ W, int k0, int ncB,
                                        unsigned int* __restrict__ dst, int tid) {
#pragma unroll
    for (int i = 0; i < 2; ++i) {
        const int id = tid + i * 256;            // 512 units
        const int kp = id >> 5;                  // k-pair 0..15
        const int fq = (id & 31) << 2;           // col 0..124 step 4
        const float4 w0 = *(const float4*)(W + (size_t)(k0 + 2 * kp)     * 1024 + ncB + fq);
        const float4 w1 = *(const float4*)(W + (size_t)(k0 + 2 * kp + 1) * 1024 + ncB + fq);
        dst[(fq + 0) * 16 + kp] = pk16(w0.x, w1.x);
        dst[(fq + 1) * 16 + kp] = pk16(w0.y, w1.y);
        dst[(fq + 2) * 16 + kp] = pk16(w0.z, w1.z);
        dst[(fq + 3) * 16 + kp] = pk16(w0.w, w1.w);
    }
}

__device__ __forceinline__ void wmma2(const v16h& a, const _Float16* Bs,
                                      int lcol, int lane, v8f* acc) {
#pragma unroll
    for (int nt = 0; nt < 2; ++nt) {
        const v16h b = load_b(Bs, lcol + nt * 16, lane);
        acc[nt] = __builtin_amdgcn_wmma_f32_16x16x32_f16(
            false, a, false, b, (short)0, acc[nt], false, false);
    }
}

// ---------------------------------------------------------------------------
// Fused per-expert MLP: H = silu(X@Wg) * (X@Wu) kept in LDS (f16),
// Y = H@Wd scaled by routing weight, scatter-added to out.
// Block: 256 threads = 8 waves, tile = 32 tokens. Wave grid 2(M) x 4(N).
// Double-buffered weight staging: one barrier per K-step, global loads of
// step k+32 overlap WMMA of step k.
// ---------------------------------------------------------------------------
__global__ void __launch_bounds__(256, 1)
moe_kernel(const float* __restrict__ X,
           const float* __restrict__ Wg,
           const float* __restrict__ Wu,
           const float* __restrict__ Wd,
           const int*   __restrict__ cursor,
           const int*   __restrict__ tokenList,
           const float* __restrict__ wList,
           float* __restrict__ out) {
    extern __shared__ char smem[];
    _Float16* Xs   = (_Float16*)(smem);             // [32][1024] f16 = 64 KB
    _Float16* Hs   = (_Float16*)(smem + 65536);     // [32][1024] f16 = 64 KB
    _Float16* Bg0  = (_Float16*)(smem + 131072);    // [128][32] f16 = 8 KB
    _Float16* Bg1  = (_Float16*)(smem + 139264);
    _Float16* Bu0  = (_Float16*)(smem + 147456);
    _Float16* Bu1  = (_Float16*)(smem + 155648);
    int*      sTok = (int*)  (smem + 163840);       // [32]
    float*    sW   = (float*)(smem + 163968);       // [32]

    const int e   = blockIdx.y;
    const int m0  = blockIdx.x * MTILE;
    const int cnt = cursor[e];
    if (m0 >= cnt) return;                          // uniform across the block

    const int tid  = threadIdx.x;
    const int lane = tid & 31;
    const int wv   = tid >> 5;
    const int wr   = wv >> 2;                       // 0..1 : M half
    const int wc   = wv & 3;                        // 0..3 : N quarter
    const int mB   = wr * 16;
    const int lcol = wc * 32;

    if (tid < MTILE) {
        const int slot = m0 + tid;
        int tok = 0; float w = 0.f;
        if (slot < cnt) { tok = tokenList[e * CAP + slot]; w = wList[e * CAP + slot]; }
        sTok[tid] = tok;  sW[tid] = w;              // padding rows get weight 0
    }
    __syncthreads();

    // ---- gather token tile X[32][1024] -> f16 LDS (packed b32 stores) ----
#pragma unroll 4
    for (int i = 0; i < 32; ++i) {
        const int id = tid + i * 256;               // 8192 float4 slots
        const int r  = id >> 8;
        const int c4 = (id & 255) << 2;
        const float4 x = *(const float4*)(X + (size_t)sTok[r] * DDIM + c4);
        unsigned int* dst = (unsigned int*)(Xs + r * DDIM + c4);
        dst[0] = pk16(x.x, x.y);
        dst[1] = pk16(x.z, x.w);
    }

    const float* Wg_e = Wg + (size_t)e * DDIM * FDIM;
    const float* Wu_e = Wu + (size_t)e * DDIM * FDIM;
    const float* Wd_e = Wd + (size_t)e * FDIM * DDIM;
    unsigned int* const g0p = (unsigned int*)Bg0;
    unsigned int* const g1p = (unsigned int*)Bg1;
    unsigned int* const u0p = (unsigned int*)Bu0;
    unsigned int* const u1p = (unsigned int*)Bu1;

    // ================= stage 1: H = silu(X@Wg) * (X@Wu) =================
#pragma unroll 1
    for (int fc = 0; fc < 8; ++fc) {
        const int fcB = fc * 128;
        v8f accG[2] = {}, accU[2] = {};

        stage_w(Wg_e, 0, fcB, g0p, tid);
        stage_w(Wu_e, 0, fcB, u0p, tid);
        __syncthreads();

#pragma unroll 1
        for (int k0 = 0; k0 < DDIM; k0 += 64) {
            // prefetch k0+32 into buf1 while computing k0 from buf0
            stage_w(Wg_e, k0 + 32, fcB, g1p, tid);
            stage_w(Wu_e, k0 + 32, fcB, u1p, tid);
            {
                const v16h a = load_a(Xs, mB, k0, lane);
                wmma2(a, Bg0, lcol, lane, accG);
                wmma2(a, Bu0, lcol, lane, accU);
            }
            __syncthreads();
            // prefetch k0+64 into buf0 while computing k0+32 from buf1
            if (k0 + 64 < DDIM) {
                stage_w(Wg_e, k0 + 64, fcB, g0p, tid);
                stage_w(Wu_e, k0 + 64, fcB, u0p, tid);
            }
            {
                const v16h a = load_a(Xs, mB, k0 + 32, lane);
                wmma2(a, Bg1, lcol, lane, accG);
                wmma2(a, Bu1, lcol, lane, accU);
            }
            __syncthreads();
        }

        // epilogue: silu(g)*u -> Hs (f16). C/D layout: row = j + 8*(lane>=16)
#pragma unroll
        for (int nt = 0; nt < 2; ++nt) {
            const int cB = fcB + lcol + nt * 16 + (lane & 15);
            const int rO = mB + ((lane >> 4) << 3);
#pragma unroll
            for (int j = 0; j < 8; ++j) {
                const float g = accG[nt][j];
                const float u = accU[nt][j];
                // silu via v_rcp_f32 (no IEEE divide sequence)
                const float h = g * __builtin_amdgcn_rcpf(1.f + __expf(-g)) * u;
                Hs[(rO + j) * FDIM + cB] = (_Float16)h;
            }
        }
    }

    // ================= stage 2: Y = H @ Wd, scaled scatter-add =================
#pragma unroll 1
    for (int dc = 0; dc < 8; ++dc) {
        const int dcB = dc * 128;
        v8f accY[2] = {};

        stage_w(Wd_e, 0, dcB, g0p, tid);
        __syncthreads();                            // also fences stage-1 Hs writes

#pragma unroll 1
        for (int k0 = 0; k0 < FDIM; k0 += 64) {
            stage_w(Wd_e, k0 + 32, dcB, g1p, tid);
            {
                const v16h a = load_a(Hs, mB, k0, lane);
                wmma2(a, Bg0, lcol, lane, accY);
            }
            __syncthreads();
            if (k0 + 64 < FDIM) stage_w(Wd_e, k0 + 64, dcB, g0p, tid);
            {
                const v16h a = load_a(Hs, mB, k0 + 32, lane);
                wmma2(a, Bg1, lcol, lane, accY);
            }
            __syncthreads();
        }

        // exactly two float adds per output element across experts -> deterministic
        {
            const int rO = mB + ((lane >> 4) << 3);
            float wt[8]; int tk[8];
#pragma unroll
            for (int j = 0; j < 8; ++j) { wt[j] = sW[rO + j]; tk[j] = sTok[rO + j]; }
#pragma unroll
            for (int nt = 0; nt < 2; ++nt) {
                const int c = dcB + lcol + nt * 16 + (lane & 15);
#pragma unroll
                for (int j = 0; j < 8; ++j) {
                    atomicAdd(out + (size_t)tk[j] * DDIM + c, accY[nt][j] * wt[j]);
                }
            }
        }
    }
}

// ---------------------------------------------------------------------------
extern "C" void kernel_launch(void* const* d_in, const int* in_sizes, int n_in,
                              void* d_out, int out_size, void* d_ws, size_t ws_size,
                              hipStream_t stream) {
    const float* X     = (const float*)d_in[0];  // hidden_states [2,2048,1024]
    const float* gateW = (const float*)d_in[1];  // [1024,8]
    const float* Wg    = (const float*)d_in[2];  // gate_proj [8,1024,1024]
    const float* Wu    = (const float*)d_in[3];  // up_proj   [8,1024,1024]
    const float* Wd    = (const float*)d_in[4];  // down_proj [8,1024,1024]

    float* out    = (float*)d_out;                       // [T*D]
    float* logits = out + (size_t)T_TOK * DDIM;          // then [T*E]

    // workspace: cursor[8] (padded to 256B) | tokenList[8*4096] | wList[8*4096]
    int*   cursor    = (int*)d_ws;
    int*   tokenList = (int*)((char*)d_ws + 256);
    float* wList     = (float*)((char*)d_ws + 256 + (size_t)NEXP * CAP * sizeof(int));

    (void)hipMemsetAsync(d_ws, 0, 256, stream);
    (void)hipMemsetAsync(d_out, 0, (size_t)T_TOK * DDIM * sizeof(float), stream);

    router_kernel<<<T_TOK / 8, 256, 0, stream>>>(X, gateW, logits, cursor, tokenList, wList);

    const int SMEM = 164096;   // 2x64KB X/H tiles + 4x8KB double-buffered stage + 256B
    (void)hipFuncSetAttribute(reinterpret_cast<const void*>(moe_kernel),
                              hipFuncAttributeMaxDynamicSharedMemorySize, SMEM);
    moe_kernel<<<dim3(T_TOK / MTILE, NEXP), 256, SMEM, stream>>>(
        X, Wg, Wu, Wd, cursor, tokenList, wList, out);
}
